// Block6_77154792505752
// MI455X (gfx1250) — matrix operand order repaired and checked by reading, not compiled, not run on hardware
//
#include <hip/hip_runtime.h>
#include <hip/hip_bf16.h>
#include <stdint.h>

typedef __bf16 bf16;
typedef __attribute__((ext_vector_type(16))) __bf16 v16bf;
typedef __attribute__((ext_vector_type(8)))  __bf16 v8bf;
typedef __attribute__((ext_vector_type(8)))  float  v8f;

#define BB 4
#define TT 4096
#define CC 2048
#define FF 7168
#define BT (BB*TT)

// ---------------------------------------------------------------------------
// Weight transpose + dequant-to-exact-bf16:  Wt[n][k] = (bf16)W_u8[k][n]
// (u8 0..255 is exact in bf16; scales/means are applied in GEMM epilogues)
// ---------------------------------------------------------------------------
__global__ __launch_bounds__(256)
void transpose_u8_bf16(const uint8_t* __restrict__ W, bf16* __restrict__ Wt,
                       const int R, const int Ncol)
{
    __shared__ float tile[32][33];
    const int tx = threadIdx.x, ty = threadIdx.y;
    const int bx = blockIdx.x * 32, by = blockIdx.y * 32;
#pragma unroll
    for (int i = 0; i < 32; i += 8)
        tile[ty + i][tx] = (float)W[(size_t)(by + ty + i) * Ncol + bx + tx];
    __syncthreads();
#pragma unroll
    for (int i = 0; i < 32; i += 8)
        Wt[(size_t)(bx + ty + i) * R + by + tx] = (bf16)tile[tx][ty + i];
}

// ---------------------------------------------------------------------------
// Prep: token shift, time-mix, fold per-row quant scale ry into activations,
// compute per-token scalars Sa=sum(x*ry), Sx=sum(x), D=sum(x*my) in f32.
// One wave per token. Also zeroes the GEMM1 atomic accumulators.
// ---------------------------------------------------------------------------
__global__ __launch_bounds__(256)
void prep_kernel(const float* __restrict__ x, const float* __restrict__ last_state,
                 const float* __restrict__ tmk, const float* __restrict__ tmr,
                 const float* __restrict__ kry, const float* __restrict__ kmy,
                 const float* __restrict__ rry, const float* __restrict__ rmy,
                 bf16* __restrict__ aK, bf16* __restrict__ aR,
                 float* __restrict__ SaK, float* __restrict__ SxK, float* __restrict__ DK,
                 float* __restrict__ SaR, float* __restrict__ SxR, float* __restrict__ DR,
                 float* __restrict__ Sa2, float* __restrict__ Sx2, float* __restrict__ D2)
{
    const int wave = threadIdx.x >> 5, lane = threadIdx.x & 31;
    const int tok = blockIdx.x * 8 + wave;
    const int b = tok / TT, ti = tok % TT;
    const float* xrow = x + (size_t)tok * CC;
    const float* prow = (ti == 0) ? (last_state + (size_t)b * CC) : (xrow - CC);

    float sak = 0.f, sxk = 0.f, dk = 0.f, sar = 0.f, sxr = 0.f, dr = 0.f;
    for (int c = lane; c < CC; c += 32) {
        const float xc = xrow[c], pc = prow[c];
        const float mk = tmk[c], mr = tmr[c];
        const float xk = pc * mk + xc * (1.f - mk);
        const float xr = pc * mr + xc * (1.f - mr);
        const float ak = xk * kry[c];
        const float ar = xr * rry[c];
        aK[(size_t)tok * CC + c] = (bf16)ak;
        aR[(size_t)tok * CC + c] = (bf16)ar;
        sak += ak; sxk += xk; dk += xk * kmy[c];
        sar += ar; sxr += xr; dr += xr * rmy[c];
    }
#pragma unroll
    for (int off = 1; off < 32; off <<= 1) {
        sak += __shfl_xor(sak, off, 32); sxk += __shfl_xor(sxk, off, 32);
        dk  += __shfl_xor(dk,  off, 32); sar += __shfl_xor(sar, off, 32);
        sxr += __shfl_xor(sxr, off, 32); dr  += __shfl_xor(dr,  off, 32);
    }
    if (lane == 0) {
        SaK[tok] = sak; SxK[tok] = sxk; DK[tok] = dk;
        SaR[tok] = sar; SxR[tok] = sxr; DR[tok] = dr;
        Sa2[tok] = 0.f; Sx2[tok] = 0.f; D2[tok] = 0.f;
    }
}

// ---------------------------------------------------------------------------
// WMMA GEMM: D[m,n] = sum_k A[m,k]*Bt[n,k]  (bf16 in, f32 acc)
//   block = 256 threads = 8 waves, tile 128(M) x 128(N) x 32(K)
//   wave (wM=wave>>1, wN=wave&1): 32 rows x 64 cols -> 2x4 WMMA tiles
//   ping-pong LDS, async memory->LDS staging (ASYNCcnt), 1 barrier / K-step,
//   K-loop unrolled x2 so buffer indices are compile-time immediates.
// MODE 1: k-path (relu^2, fold vry, emit bf16 A-matrix for GEMM2 + F-sums)
// MODE 2: kv-path (write f32 kv into fOut)
// MODE 3: r-path  (sigmoid, multiply fOut in place)
// ---------------------------------------------------------------------------
#define BKP 40   // padded LDS row stride in halves (80B: 16B-aligned, bank-spread)

template <int MODE>
__global__ __launch_bounds__(256)
void gemm_wmma(const bf16* __restrict__ A, const bf16* __restrict__ Bt,
               const int K,
               const float* __restrict__ colScale, const float* __restrict__ colMean,
               const float* __restrict__ rowSa, const float* __restrict__ rowSx,
               const float* __restrict__ rowD,
               bf16* __restrict__ kbOut, const float* __restrict__ vry_p,
               const float* __restrict__ vmy_p,
               float* __restrict__ accSa, float* __restrict__ accSx,
               float* __restrict__ accD, float* __restrict__ fOut)
{
    __shared__ __align__(16) bf16 sA[2][128 * BKP];
    __shared__ __align__(16) bf16 sB[2][128 * BKP];

    const int tid = threadIdx.x;
    const int wave = tid >> 5, lane = tid & 31, l16 = lane & 15, half = lane >> 4;
    const int waveM = wave >> 1, waveN = wave & 1;
    const size_t mBase = (size_t)blockIdx.y * 128;
    const size_t nBase = (size_t)blockIdx.x * 128;

    // cooperative staging: 2 threads per 64B tile row, 32B (=2 x b128) each
    const int sRow = tid >> 1, sPart = tid & 1;
    const bf16* aG = A  + (mBase + sRow) * (size_t)K + sPart * 16;
    const bf16* bG = Bt + (nBase + sRow) * (size_t)K + sPart * 16;
    const uint32_t bufStride = (uint32_t)(128 * BKP * 2);
    const uint32_t aL0 = (uint32_t)(uintptr_t)(&sA[0][0]) +
                         (uint32_t)(sRow * BKP + sPart * 16) * 2u;
    const uint32_t bL0 = (uint32_t)(uintptr_t)(&sB[0][0]) +
                         (uint32_t)(sRow * BKP + sPart * 16) * 2u;

    // async memory->LDS stage of K-slab kb into buffer BUF (compile-time)
    auto issue = [&](int kb, uint32_t bufOff) {
        const bf16* ag = aG + (size_t)kb * 32;
        const bf16* bg = bG + (size_t)kb * 32;
        const uint32_t al = aL0 + bufOff;
        const uint32_t bl = bL0 + bufOff;
        // INST_OFFSET applies to both LDS and global addresses (ISA 15.18.3)
        asm volatile("global_load_async_to_lds_b128 %0, %1, off\n\t"
                     "global_load_async_to_lds_b128 %0, %1, off offset:16"
                     :: "v"(al), "v"((uint64_t)(uintptr_t)ag) : "memory");
        asm volatile("global_load_async_to_lds_b128 %0, %1, off\n\t"
                     "global_load_async_to_lds_b128 %0, %1, off offset:16"
                     :: "v"(bl), "v"((uint64_t)(uintptr_t)bg) : "memory");
    };

    v8f acc[2][4] = {};

    auto compute = [&](const bf16* sAc, const bf16* sBc) {
        // B fragments (K-contig per lane: half 0 -> K 0..15, half 1 -> K 16..31)
        v16bf bfr[4];
#pragma unroll
        for (int t = 0; t < 4; ++t)
            bfr[t] = *(const v16bf*)(sBc + (waveN * 64 + t * 16 + l16) * BKP + half * 16);
#pragma unroll
        for (int i = 0; i < 2; ++i) {
            // A fragment (16-bit A 16x32): lanes 0-15 K 0..15, 16-31 K 16..31 (matches B)
            const bf16* ap = sAc + (waveM * 32 + i * 16 + l16) * BKP + half * 16;
            v8bf a0 = *(const v8bf*)ap;
            v8bf a1 = *(const v8bf*)(ap + 8);
            v16bf af = __builtin_shufflevector(a0, a1, 0, 1, 2, 3, 4, 5, 6, 7,
                                               8, 9, 10, 11, 12, 13, 14, 15);
#pragma unroll
            for (int t = 0; t < 4; ++t)
                acc[i][t] = __builtin_amdgcn_wmma_f32_16x16x32_bf16(
                    false, af, false, bfr[t], (short)0, acc[i][t], false, false);
        }
    };

    const int nk = K >> 5;            // always even (K = 2048 or 7168)
    issue(0, 0u);

    int kb = 0;
    for (; kb + 2 < nk; kb += 2) {    // steady state: compile-time ping-pong
        asm volatile("s_wait_asynccnt 0x0" ::: "memory");
        __syncthreads();
        issue(kb + 1, bufStride);
        compute(sA[0], sB[0]);

        asm volatile("s_wait_asynccnt 0x0" ::: "memory");
        __syncthreads();
        issue(kb + 2, 0u);
        compute(sA[1], sB[1]);
    }
    // peeled tail: kb == nk-2
    asm volatile("s_wait_asynccnt 0x0" ::: "memory");
    __syncthreads();
    issue(nk - 1, bufStride);
    compute(sA[0], sB[0]);

    asm volatile("s_wait_asynccnt 0x0" ::: "memory");
    __syncthreads();
    compute(sA[1], sB[1]);

    // epilogue: C/D layout -- VGPR j: lanes 0-15 -> (M=j, N=l16), 16-31 -> (M=j+8)
#pragma unroll
    for (int i = 0; i < 2; ++i) {
#pragma unroll
        for (int j = 0; j < 8; ++j) {
            const size_t m = mBase + waveM * 32 + i * 16 + j + half * 8;
            const float sa = rowSa[m] * 0.5f;   // folds the (w+0.5) half-LSB term
            const float sx = rowSx[m];
            const float dd = rowD[m];
            float pk = 0.f, ps = 0.f, pd = 0.f;
#pragma unroll
            for (int t = 0; t < 4; ++t) {
                const size_t n = nBase + waveN * 64 + t * 16 + l16;
                float v = colScale[n] * (acc[i][t][j] + sa) + dd + colMean[n] * sx;
                if (MODE == 1) {
                    v = v > 0.f ? v : 0.f;
                    v = v * v;                          // relu^2
                    const float kb2 = v * vry_p[n];     // fold next GEMM's row scale
                    kbOut[m * (size_t)FF + n] = (bf16)kb2;
                    pk += kb2; ps += v; pd += v * vmy_p[n];
                } else if (MODE == 2) {
                    fOut[m * (size_t)CC + n] = v;       // kv
                } else {
                    const float r = 1.0f / (1.0f + __expf(-v));
                    fOut[m * (size_t)CC + n] *= r;      // r * kv, in place
                }
            }
            if (MODE == 1) {
#pragma unroll
                for (int off = 1; off < 16; off <<= 1) {
                    pk += __shfl_xor(pk, off, 32);
                    ps += __shfl_xor(ps, off, 32);
                    pd += __shfl_xor(pd, off, 32);
                }
                if (l16 == 0) {
                    atomicAdd(&accSa[m], pk);
                    atomicAdd(&accSx[m], ps);
                    atomicAdd(&accD[m], pd);
                }
            }
        }
    }
}

// new_state = x[:, -1, :]
__global__ __launch_bounds__(256)
void tail_kernel(const float* __restrict__ x, float* __restrict__ outTail)
{
    const int i = blockIdx.x * 256 + threadIdx.x;   // 0 .. B*C-1
    const int b = i / CC, c = i % CC;
    outTail[i] = x[((size_t)b * TT + TT - 1) * CC + c];
}

extern "C" void kernel_launch(void* const* d_in, const int* in_sizes, int n_in,
                              void* d_out, int out_size, void* d_ws, size_t ws_size,
                              hipStream_t stream)
{
    const float* x          = (const float*)d_in[0];
    const float* last_state = (const float*)d_in[1];
    const float* tmk        = (const float*)d_in[2];
    const float* tmr        = (const float*)d_in[3];
    const float* kmx        = (const float*)d_in[4];
    const float* krx        = (const float*)d_in[5];
    const float* kmy        = (const float*)d_in[6];
    const float* kry        = (const float*)d_in[7];
    const float* vmx        = (const float*)d_in[8];
    const float* vrx        = (const float*)d_in[9];
    const float* vmy        = (const float*)d_in[10];
    const float* vry        = (const float*)d_in[11];
    const float* rmx        = (const float*)d_in[12];
    const float* rrx        = (const float*)d_in[13];
    const float* rmy        = (const float*)d_in[14];
    const float* rry        = (const float*)d_in[15];
    const uint8_t* wk       = (const uint8_t*)d_in[16];
    const uint8_t* wv       = (const uint8_t*)d_in[17];
    const uint8_t* wr       = (const uint8_t*)d_in[18];
    float* out = (float*)d_out;

    char* ws = (char*)d_ws;
    size_t off = 0;
    auto alloc = [&](size_t bytes) -> void* {
        void* p = ws + off;
        off += (bytes + 255) & ~(size_t)255;
        return p;
    };
    bf16* aK   = (bf16*)alloc((size_t)BT * CC * 2);
    bf16* aR   = (bf16*)alloc((size_t)BT * CC * 2);
    bf16* wkT  = (bf16*)alloc((size_t)FF * CC * 2);  // [f][c]
    bf16* wvT  = (bf16*)alloc((size_t)CC * FF * 2);  // [c][f]
    bf16* wrT  = (bf16*)alloc((size_t)CC * CC * 2);  // [d][c]
    bf16* kbM  = (bf16*)alloc((size_t)BT * FF * 2);  // GEMM2 A-matrix
    float* SaK = (float*)alloc((size_t)BT * 4);
    float* SxK = (float*)alloc((size_t)BT * 4);
    float* DK  = (float*)alloc((size_t)BT * 4);
    float* SaR = (float*)alloc((size_t)BT * 4);
    float* SxR = (float*)alloc((size_t)BT * 4);
    float* DR  = (float*)alloc((size_t)BT * 4);
    float* Sa2 = (float*)alloc((size_t)BT * 4);
    float* Sx2 = (float*)alloc((size_t)BT * 4);
    float* D2  = (float*)alloc((size_t)BT * 4);

    dim3 tb(32, 8);
    transpose_u8_bf16<<<dim3(FF / 32, CC / 32), tb, 0, stream>>>(wk, wkT, CC, FF);
    transpose_u8_bf16<<<dim3(CC / 32, FF / 32), tb, 0, stream>>>(wv, wvT, FF, CC);
    transpose_u8_bf16<<<dim3(CC / 32, CC / 32), tb, 0, stream>>>(wr, wrT, CC, CC);

    prep_kernel<<<BT / 8, 256, 0, stream>>>(x, last_state, tmk, tmr, kry, kmy,
                                            rry, rmy, aK, aR,
                                            SaK, SxK, DK, SaR, SxR, DR,
                                            Sa2, Sx2, D2);

    gemm_wmma<1><<<dim3(FF / 128, BT / 128), 256, 0, stream>>>(
        aK, wkT, CC, krx, kmx, SaK, SxK, DK,
        kbM, vry, vmy, Sa2, Sx2, D2, nullptr);

    gemm_wmma<2><<<dim3(CC / 128, BT / 128), 256, 0, stream>>>(
        kbM, wvT, FF, vrx, vmx, Sa2, Sx2, D2,
        nullptr, nullptr, nullptr, nullptr, nullptr, nullptr, out);

    gemm_wmma<3><<<dim3(CC / 128, BT / 128), 256, 0, stream>>>(
        aR, wrT, CC, rrx, rmx, SaR, SxR, DR,
        nullptr, nullptr, nullptr, nullptr, nullptr, nullptr, out);

    tail_kernel<<<(BB * CC) / 256, 256, 0, stream>>>(x, out + (size_t)BT * CC);
}
